// NN_mlpc_qc_1460288881269
// MI455X (gfx1250) — matrix operand order repaired and checked by reading, not compiled, not run on hardware
//
#include <hip/hip_runtime.h>
#include <hip/hip_bf16.h>
#include <cstdint>

// ---------------- constants from the reference ----------------
#define BATCH 2048
#define REPR  8192
#define LOWD  32
#define FDIM  256
#define HDIM  512
#define LEVELS 3
#define DDIM  8
#define BINS  16
#define SDIM  (LEVELS * DDIM)        // 24
#define BS    (BATCH * SDIM)         // 49152
#define HEADN (SDIM * BINS)          // 384

typedef __attribute__((ext_vector_type(16))) __bf16 v16bf;
typedef __attribute__((ext_vector_type(8)))  __bf16 v8bf;
typedef __attribute__((ext_vector_type(8)))  float  v8f;
typedef __attribute__((ext_vector_type(4)))  int    v4i;

#define GLOBAL_AS __attribute__((address_space(1)))
#define LDS_AS    __attribute__((address_space(3)))

#if __has_builtin(__builtin_amdgcn_global_load_async_to_lds_b128) && \
    __has_builtin(__builtin_amdgcn_s_wait_asynccnt)
#define USE_ASYNC_LDS 1
#else
#define USE_ASYNC_LDS 0
#endif

#if USE_ASYNC_LDS
// builtin parameter types are <4 x i32> pointers in AS1 (src) / AS3 (dst)
__device__ __forceinline__ void async_copy16(const void* gsrc, void* ldst) {
    __builtin_amdgcn_global_load_async_to_lds_b128(
        (GLOBAL_AS v4i*)((GLOBAL_AS void*)gsrc),
        (LDS_AS  v4i*)((LDS_AS  void*)ldst),
        0, 0);
}
#endif

// ---------------- helpers ----------------
__device__ __forceinline__ unsigned short f2bf(float f) {
    union { float f; unsigned int u; } v; v.f = f;
    unsigned int u = v.u;
    unsigned int r = u + 0x7fffu + ((u >> 16) & 1u);   // RNE
    return (unsigned short)(r >> 16);
}
__device__ __forceinline__ float bf2f(unsigned short h) {
    union { unsigned int u; float f; } v; v.u = ((unsigned int)h) << 16;
    return v.f;
}

// ---------------- elementwise f32 -> bf16 ----------------
__global__ void k_f32_to_bf16(const float* __restrict__ src,
                              unsigned short* __restrict__ dst, int n) {
    int i = blockIdx.x * blockDim.x + threadIdx.x;
    if (i < n) dst[i] = f2bf(src[i]);
}

// ---------------- transpose (K x N f32) -> (N x K bf16) ----------------
__global__ void k_transpose_bf16(const float* __restrict__ src,
                                 unsigned short* __restrict__ dst,
                                 int K, int N) {
    int i = blockIdx.x * blockDim.x + threadIdx.x;
    if (i >= K * N) return;
    int k = i / N, n = i % N;
    dst[(size_t)n * K + k] = f2bf(src[i]);
}

// ---------------- low-obs tiny GEMM: pre_low (B x 512) ----------------
__global__ void k_low_gemm(const float* __restrict__ low,
                           const float* __restrict__ vW,   // 32x256
                           const float* __restrict__ qW,   // 32x256
                           float* __restrict__ out) {
    int t = blockIdx.x * blockDim.x + threadIdx.x;
    if (t >= BATCH * 2 * FDIM) return;
    int b = t >> 9;            // /512
    int j = t & 511;
    const float* W = (j < FDIM) ? vW : qW;
    int jj = (j < FDIM) ? j : j - FDIM;
    const float* lb = low + (size_t)b * LOWD;
    float s = 0.f;
#pragma unroll
    for (int k = 0; k < LOWD; ++k) s += lb[k] * W[k * FDIM + jj];
    out[(size_t)b * (2 * FDIM) + j] = s;
}

// ---------------- WMMA bf16 GEMM with LDS-staged B ----------------
// C(MxN f32) = A(MxK) * Bt(NxK)^T
// block = 256 threads (8 waves); block tile 128(M) x 64(N); wave tile 16 x 64.
// B slab (64 x 32 bf16 = 4 KB) staged in LDS once per block per K-step,
// double-buffered via gfx1250 GLOBAL_LOAD_ASYNC_TO_LDS_B128 (ASYNCcnt).
// Main loop is peeled so the steady state is branch-free:
//   issue(next) ; s_wait_asynccnt 1 ; barrier ; 4x wmma ; barrier
// LDS row stride padded to 40 elements (20 words) -> conflict-free ds_load_b128.
#define BROW 40                       // padded row stride (elements)
#define BBUF (64 * BROW)              // one buffer, elements (2560 -> 5120 B)

__device__ __forceinline__ void gemm_compute_step(
        const __bf16* __restrict__ Arow, int k0,
        const __bf16* __restrict__ bbase, int ml, int hi, v8f acc[4]) {
    // A fragment: lanes 0-15 -> K[0..7],[16..23]; lanes 16-31 -> K[8..15],[24..31]
    v8bf alo = *(const v8bf*)(Arow + k0);
    v8bf ahi = *(const v8bf*)(Arow + k0 + 16);
    v16bf a = __builtin_shufflevector(alo, ahi,
                0,1,2,3,4,5,6,7,8,9,10,11,12,13,14,15);
#pragma unroll
    for (int t = 0; t < 4; ++t) {
        // B fragment: col = ml; lanes 0-15 -> K[0..7],[16..23]; lanes 16-31 -> K[8..15],[24..31]
        const __bf16* bp = bbase + (t * 16 + ml) * BROW + hi * 8;
        v8bf blo  = *(const v8bf*)(bp);
        v8bf bhi2 = *(const v8bf*)(bp + 16);
        v16bf b = __builtin_shufflevector(blo, bhi2,
                    0,1,2,3,4,5,6,7,8,9,10,11,12,13,14,15);
        acc[t] = __builtin_amdgcn_wmma_f32_16x16x32_bf16(
                    false, a, false, b, (short)0, acc[t], false, false);
    }
}

__global__ __launch_bounds__(256) void k_gemm_bf16(
        const __bf16* __restrict__ A, int lda,
        const __bf16* __restrict__ Bt, int ldb,
        float* __restrict__ C, int ldc, int K) {
    __shared__ __bf16 smem[2 * BBUF];          // 10240 bytes

    const int tid  = threadIdx.x;
    const int wave = tid >> 5;
    const int lane = tid & 31;
    const int ml   = lane & 15;
    const int hi   = lane >> 4;
    const int m0   = blockIdx.y * 128 + wave * 16;
    const int n0   = blockIdx.x * 64;

    // this thread's 16-byte B chunk: row 0..63 (= n-local), seg 0..3 (= 8 k-elems)
    const int brow = tid >> 2;
    const int bseg = tid & 3;
    const __bf16* gB = Bt + (size_t)(n0 + brow) * ldb + bseg * 8;
    __bf16* lB = &smem[brow * BROW + bseg * 8];

    v8f acc[4] = {};
    const __bf16* Arow = A + (size_t)(m0 + ml) * lda + hi * 8;

    const int nk = K / 32;
    int buf = 0;

    // prologue: stage slab 0
#if USE_ASYNC_LDS
    async_copy16(gB, lB);
#else
    *(v8bf*)lB = *(const v8bf*)gB;
#endif

    // main loop: iterations 0 .. nk-2, always prefetch (branch-free body)
    int kk = 0;
    for (; kk < nk - 1; ++kk) {
        const int k0 = kk * 32;
#if USE_ASYNC_LDS
        async_copy16(gB + (size_t)(k0 + 32), lB + (buf ^ 1) * BBUF);
        __builtin_amdgcn_s_wait_asynccnt(1);   // current slab landed
#else
        *(v8bf*)(lB + (buf ^ 1) * BBUF) = *(const v8bf*)(gB + (size_t)(k0 + 32));
#endif
        __syncthreads();   // slab `buf` visible to all waves
        gemm_compute_step(Arow, k0, &smem[buf * BBUF], ml, hi, acc);
        __syncthreads();   // all reads of slab `buf` done before it is overwritten
        buf ^= 1;
    }

    // epilogue: last slab
#if USE_ASYNC_LDS
    __builtin_amdgcn_s_wait_asynccnt(0);
#endif
    __syncthreads();
    gemm_compute_step(Arow, kk * 32, &smem[buf * BBUF], ml, hi, acc);

    // D layout: row = v + 8*hi, col = ml
#pragma unroll
    for (int t = 0; t < 4; ++t)
#pragma unroll
        for (int v = 0; v < 8; ++v)
            C[(size_t)(m0 + v + 8 * hi) * ldc + n0 + t * 16 + ml] = acc[t][v];
}

// ---------------- LayerNorm + activation, f32 in -> bf16 out ----------------
// one block per row; width = 256 or 512; act: 0 = tanh, 1 = silu
__global__ __launch_bounds__(256) void k_ln_act(
        const float* __restrict__ src, int srcStride, int width,
        const float* __restrict__ g, const float* __restrict__ bb, int act,
        unsigned short* __restrict__ dst, int dstStride) {
    const int row = blockIdx.x;
    const int tid = threadIdx.x;
    const float* x = src + (size_t)row * srcStride;

    float v0 = x[tid], v1 = 0.f;
    float s = v0, s2 = v0 * v0;
    if (width == 512) { v1 = x[tid + 256]; s += v1; s2 += v1 * v1; }

    __shared__ float r1[256], r2[256];
    r1[tid] = s; r2[tid] = s2;
    __syncthreads();
    for (int off = 128; off > 0; off >>= 1) {
        if (tid < off) { r1[tid] += r1[tid + off]; r2[tid] += r2[tid + off]; }
        __syncthreads();
    }
    const float inv = 1.0f / (float)width;
    const float mu  = r1[0] * inv;
    const float var = r2[0] * inv - mu * mu;
    const float rs  = rsqrtf(var + 1e-5f);

    float y = (v0 - mu) * rs * g[tid] + bb[tid];
    y = act ? (y / (1.f + __expf(-y))) : tanhf(y);
    dst[(size_t)row * dstStride + tid] = f2bf(y);
    if (width == 512) {
        float z = (v1 - mu) * rs * g[tid + 256] + bb[tid + 256];
        z = act ? (z / (1.f + __expf(-z))) : tanhf(z);
        dst[(size_t)row * dstStride + tid + 256] = f2bf(z);
    }
}

// ---------------- actions from category ----------------
// one thread per (b, d)
__global__ void k_actions(const int* __restrict__ cat, float* __restrict__ act) {
    int t = blockIdx.x * blockDim.x + threadIdx.x;
    if (t >= BATCH * DDIM) return;
    int b = t >> 3, d = t & 7;
    float lo = -1.f, hiv = 1.f;
    float M[LEVELS + 1];
    M[0] = 0.f;
#pragma unroll
    for (int l = 0; l < LEVELS; ++l) {
        float c = (float)cat[((size_t)b * LEVELS + l) * DDIM + d];
        float w = (hiv - lo) * (1.0f / BINS);
        float nl = lo + c * w;
        float nh = nl + w;
        M[l + 1] = 0.5f * (nl + nh);
        lo = nl; hiv = nh;
    }
#pragma unroll
    for (int s = 0; s < SDIM; ++s) {
        int ls = s >> 3, ds = s & 7;
        int lvl = ls + ((d < ds) ? 1 : 0);
        act[((size_t)b * SDIM + s) * DDIM + d] = M[lvl];
    }
}

// ---------------- q layer1: U + actions@W1c, LN, SiLU -> bf16 Hq ----------------
// one block per (b,s) row of 512
__global__ __launch_bounds__(256) void k_q1_ln_silu(
        const float* __restrict__ U, const float* __restrict__ act,
        const float* __restrict__ qW1,      // (520 x 512) f32
        const float* __restrict__ g, const float* __restrict__ bb,
        unsigned short* __restrict__ Hq) {
    const int bs = blockIdx.x;
    const int b  = bs / SDIM;
    const int tid = threadIdx.x;

    __shared__ float av[DDIM];
    if (tid < DDIM) av[tid] = act[(size_t)bs * DDIM + tid];
    __syncthreads();

    const float* Ub = U + (size_t)b * HDIM;
    float vv[2], s = 0.f, s2 = 0.f;
#pragma unroll
    for (int e = 0; e < 2; ++e) {
        int c = tid + e * 256;
        float val = Ub[c];
#pragma unroll
        for (int d = 0; d < DDIM; ++d)
            val += av[d] * qW1[(size_t)(2 * FDIM + d) * HDIM + c];
        vv[e] = val; s += val; s2 += val * val;
    }
    __shared__ float r1[256], r2[256];
    r1[tid] = s; r2[tid] = s2;
    __syncthreads();
    for (int off = 128; off > 0; off >>= 1) {
        if (tid < off) { r1[tid] += r1[tid + off]; r2[tid] += r2[tid + off]; }
        __syncthreads();
    }
    const float inv = 1.0f / (float)HDIM;
    const float mu  = r1[0] * inv;
    const float var = r2[0] * inv - mu * mu;
    const float rs  = rsqrtf(var + 1e-5f);
#pragma unroll
    for (int e = 0; e < 2; ++e) {
        int c = tid + e * 256;
        float y = (vv[e] - mu) * rs * g[c] + bb[c];
        y = y / (1.f + __expf(-y));                 // silu
        Hq[(size_t)bs * HDIM + c] = f2bf(y);
    }
}

// ---------------- value head ----------------
__global__ void k_value_head(const unsigned short* __restrict__ hv,
                             const float* __restrict__ Wh,
                             const float* __restrict__ bh,
                             float* __restrict__ out) {
    int b = blockIdx.x * blockDim.x + threadIdx.x;
    if (b >= BATCH) return;
    const unsigned short* r = hv + (size_t)b * HDIM;
    float s = 0.f;
    for (int n = 0; n < HDIM; ++n) s += bf2f(r[n]) * Wh[n];
    out[b] = s + bh[0];
}

// ---------------- adv head (diagonal only) ----------------
// thread per (bs, j): out = Hq[bs,:] @ q_Wh[:, s*16+j] + q_bh[s*16+j]
__global__ void k_adv_head(const unsigned short* __restrict__ Hq,
                           const float* __restrict__ Wh,   // (512 x 384)
                           const float* __restrict__ bh,   // (384)
                           float* __restrict__ out) {
    int t = blockIdx.x * blockDim.x + threadIdx.x;
    if (t >= BS * BINS) return;
    int j  = t & (BINS - 1);
    int bs = t >> 4;
    int s  = bs % SDIM;
    int col = s * BINS + j;
    const unsigned short* r = Hq + (size_t)bs * HDIM;
    float acc = bh[col];
    for (int n = 0; n < HDIM; ++n)
        acc += bf2f(r[n]) * Wh[(size_t)n * HEADN + col];
    out[(size_t)bs * BINS + j] = acc;
}

// =====================================================================
extern "C" void kernel_launch(void* const* d_in, const int* in_sizes, int n_in,
                              void* d_out, int out_size, void* d_ws, size_t ws_size,
                              hipStream_t stream) {
    (void)in_sizes; (void)n_in; (void)out_size; (void)ws_size;

    const float* rgb     = (const float*)d_in[0];
    const float* low     = (const float*)d_in[1];
    const int*   cat     = (const int*)  d_in[2];
    const float* v_W_rgb = (const float*)d_in[3];
    const float* v_g_rgb = (const float*)d_in[4];
    const float* v_b_rgb = (const float*)d_in[5];
    const float* v_W_low = (const float*)d_in[6];
    const float* v_g_low = (const float*)d_in[7];
    const float* v_b_low = (const float*)d_in[8];
    const float* v_W1    = (const float*)d_in[9];
    const float* v_g1    = (const float*)d_in[10];
    const float* v_b1    = (const float*)d_in[11];
    const float* v_W2    = (const float*)d_in[12];
    const float* v_g2    = (const float*)d_in[13];
    const float* v_b2    = (const float*)d_in[14];
    const float* v_Wh    = (const float*)d_in[15];
    const float* v_bh    = (const float*)d_in[16];
    const float* q_W_rgb = (const float*)d_in[17];
    const float* q_g_rgb = (const float*)d_in[18];
    const float* q_b_rgb = (const float*)d_in[19];
    const float* q_W_low = (const float*)d_in[20];
    const float* q_g_low = (const float*)d_in[21];
    const float* q_b_low = (const float*)d_in[22];
    const float* q_W1    = (const float*)d_in[23];
    const float* q_g1    = (const float*)d_in[24];
    const float* q_b1    = (const float*)d_in[25];
    const float* q_W2    = (const float*)d_in[26];
    const float* q_g2    = (const float*)d_in[27];
    const float* q_b2    = (const float*)d_in[28];
    const float* q_Wh    = (const float*)d_in[29];
    const float* q_bh    = (const float*)d_in[30];

    float* out_value = (float*)d_out;              // [0, 2048)
    float* out_adv   = (float*)d_out + BATCH;      // [2048, 2048 + B*S*BINS)

    // -------- workspace layout (bytes) --------
    char* ws = (char*)d_ws;
    const size_t o_X       = 0;                                   // 2048*8192*2  = 33,554,432
    const size_t o_Wtrgb   = o_X     + (size_t)BATCH * REPR * 2;  // 512*8192*2   =  8,388,608
    const size_t o_vW1t    = o_Wtrgb + (size_t)(2 * FDIM) * REPR * 2;
    const size_t o_vW2t    = o_vW1t  + (size_t)HDIM * HDIM * 2;
    const size_t o_qW1abt  = o_vW2t  + (size_t)HDIM * HDIM * 2;
    const size_t o_qW2t    = o_qW1abt+ (size_t)HDIM * HDIM * 2;
    const size_t o_prergb  = o_qW2t  + (size_t)HDIM * HDIM * 2;   // 2048*512*4
    const size_t o_prelow  = o_prergb+ (size_t)BATCH * 512 * 4;   // 2048*512*4
    const size_t o_hv      = o_prelow+ (size_t)BATCH * 512 * 4;   // 2048*512*2 bf16
    const size_t o_hq      = o_hv    + (size_t)BATCH * 512 * 2;
    const size_t o_mid     = o_hq    + (size_t)BATCH * 512 * 2;   // 2048*512*4 (pre / U)
    const size_t o_act     = o_mid   + (size_t)BATCH * 512 * 4;   // B*S*D*4
    const size_t o_Hq      = o_act   + (size_t)BS * DDIM * 4;     // 49152*512*2 bf16
    // G5 chunk f32 buffer overlays the (then dead) X region: 16384*512*4 = 33.5 MB
    const size_t o_prebig  = o_X;

    unsigned short* Xbf    = (unsigned short*)(ws + o_X);
    unsigned short* Wtrgb  = (unsigned short*)(ws + o_Wtrgb);
    unsigned short* vW1t   = (unsigned short*)(ws + o_vW1t);
    unsigned short* vW2t   = (unsigned short*)(ws + o_vW2t);
    unsigned short* qW1abt = (unsigned short*)(ws + o_qW1abt);
    unsigned short* qW2t   = (unsigned short*)(ws + o_qW2t);
    float*          prergb = (float*)(ws + o_prergb);
    float*          prelow = (float*)(ws + o_prelow);
    unsigned short* hv     = (unsigned short*)(ws + o_hv);
    unsigned short* hq     = (unsigned short*)(ws + o_hq);
    float*          mid    = (float*)(ws + o_mid);
    float*          act    = (float*)(ws + o_act);
    unsigned short* Hq     = (unsigned short*)(ws + o_Hq);
    float*          prebig = (float*)(ws + o_prebig);

    // -------- 1. conversions / transposes --------
    {
        int n = BATCH * REPR;
        k_f32_to_bf16<<<(n + 255) / 256, 256, 0, stream>>>(rgb, Xbf, n);
    }
    {
        int n = REPR * FDIM;
        k_transpose_bf16<<<(n + 255) / 256, 256, 0, stream>>>(v_W_rgb, Wtrgb, REPR, FDIM);
        k_transpose_bf16<<<(n + 255) / 256, 256, 0, stream>>>(q_W_rgb, Wtrgb + (size_t)FDIM * REPR, REPR, FDIM);
    }
    {
        int n = HDIM * HDIM;
        k_transpose_bf16<<<(n + 255) / 256, 256, 0, stream>>>(v_W1, vW1t,   HDIM, HDIM);
        k_transpose_bf16<<<(n + 255) / 256, 256, 0, stream>>>(v_W2, vW2t,   HDIM, HDIM);
        k_transpose_bf16<<<(n + 255) / 256, 256, 0, stream>>>(q_W1, qW1abt, HDIM, HDIM); // rows 0..511 only
        k_transpose_bf16<<<(n + 255) / 256, 256, 0, stream>>>(q_W2, qW2t,   HDIM, HDIM);
    }

    // -------- 2. low branch + big rgb GEMM (v & q fused, N = 512) --------
    {
        int n = BATCH * 512;
        k_low_gemm<<<(n + 255) / 256, 256, 0, stream>>>(low, v_W_low, q_W_low, prelow);
    }
    {
        dim3 grid(512 / 64, BATCH / 128);
        k_gemm_bf16<<<grid, 256, 0, stream>>>((const __bf16*)Xbf, REPR,
                                              (const __bf16*)Wtrgb, REPR,
                                              prergb, 512, REPR);
    }

    // -------- 3. LN + tanh -> hv_cat, hq_cat (bf16, B x 512 each) --------
    k_ln_act<<<BATCH, 256, 0, stream>>>(prergb,        512, 256, v_g_rgb, v_b_rgb, 0, hv,        512);
    k_ln_act<<<BATCH, 256, 0, stream>>>(prergb + 256,  512, 256, q_g_rgb, q_b_rgb, 0, hq,        512);
    k_ln_act<<<BATCH, 256, 0, stream>>>(prelow,        512, 256, v_g_low, v_b_low, 0, hv + 256,  512);
    k_ln_act<<<BATCH, 256, 0, stream>>>(prelow + 256,  512, 256, q_g_low, q_b_low, 0, hq + 256,  512);

    // -------- 4. value trunk --------
    {
        dim3 grid(512 / 64, BATCH / 128);
        k_gemm_bf16<<<grid, 256, 0, stream>>>((const __bf16*)hv, HDIM,
                                              (const __bf16*)vW1t, HDIM, mid, HDIM, HDIM);
        k_ln_act<<<BATCH, 256, 0, stream>>>(mid, HDIM, HDIM, v_g1, v_b1, 1, hv, HDIM);
        k_gemm_bf16<<<grid, 256, 0, stream>>>((const __bf16*)hv, HDIM,
                                              (const __bf16*)vW2t, HDIM, mid, HDIM, HDIM);
        k_ln_act<<<BATCH, 256, 0, stream>>>(mid, HDIM, HDIM, v_g2, v_b2, 1, hv, HDIM);
        k_value_head<<<(BATCH + 255) / 256, 256, 0, stream>>>(hv, v_Wh, v_bh, out_value);
    }

    // -------- 5. q trunk: U = [h1|h2] @ q_W1[0:512]  (per-batch, not per-s!) --------
    {
        dim3 grid(512 / 64, BATCH / 128);
        k_gemm_bf16<<<grid, 256, 0, stream>>>((const __bf16*)hq, HDIM,
                                              (const __bf16*)qW1abt, HDIM, mid, HDIM, HDIM);
    }
    {
        int n = BATCH * DDIM;
        k_actions<<<(n + 255) / 256, 256, 0, stream>>>(cat, act);
    }
    k_q1_ln_silu<<<BS, 256, 0, stream>>>(mid, act, q_W1, q_g1, q_b1, Hq);

    // -------- 6. big q GEMM (49152 x 512 x 512), chunked over M; LN+SiLU in place --------
    const int MC = 16384;
    for (int c = 0; c < BS / MC; ++c) {
        const __bf16* Achunk = (const __bf16*)(Hq + (size_t)c * MC * HDIM);
        dim3 grid(512 / 64, MC / 128);
        k_gemm_bf16<<<grid, 256, 0, stream>>>(Achunk, HDIM,
                                              (const __bf16*)qW2t, HDIM, prebig, HDIM, HDIM);
        k_ln_act<<<MC, 256, 0, stream>>>(prebig, HDIM, HDIM, q_g2, q_b2, 1,
                                         Hq + (size_t)c * MC * HDIM, HDIM);
    }

    // -------- 7. diagonal adv head --------
    {
        int n = BS * BINS;
        k_adv_head<<<(n + 255) / 256, 256, 0, stream>>>(Hq, q_Wh, q_bh, out_adv);
    }
}
